// CausalSelfAttention_8761733283918
// MI455X (gfx1250) — matrix operand order repaired and checked by reading, not compiled, not run on hardware
//
#include <hip/hip_runtime.h>
#include <hip/hip_bf16.h>

typedef __attribute__((ext_vector_type(16))) __bf16 v16bf;
typedef __attribute__((ext_vector_type(8)))  float  v8f;

#define DIM   2048
#define HEADS 16
#define KVH   4
#define HD    128
#define KVD   (KVH * HD)   // 512
#define SEQ   2048
#define BATCH 2
#define MROWS (BATCH * SEQ) // 4096

// ---------------------------------------------------------------------------
// CDNA5 async global->LDS copy helpers (ASYNCcnt-tracked, no VGPR staging)
// ---------------------------------------------------------------------------
__device__ __forceinline__ void async_copy_b128(unsigned lds_byte_off,
                                                const void* gptr) {
  asm volatile("global_load_async_to_lds_b128 %0, %1, off"
               :: "v"(lds_byte_off),
                  "v"((unsigned long long)(size_t)gptr)
               : "memory");
}
__device__ __forceinline__ void wait_async0() {
  asm volatile("s_wait_asynccnt 0x0" ::: "memory");
}

// ---------------------------------------------------------------------------
// f32 -> bf16 cast (grid-stride)
// ---------------------------------------------------------------------------
__global__ __launch_bounds__(256)
void cast_f32_bf16(const float* __restrict__ s, __bf16* __restrict__ d, long n) {
  long i = (long)blockIdx.x * blockDim.x + threadIdx.x;
  long stride = (long)gridDim.x * blockDim.x;
  for (; i < n; i += stride) d[i] = (__bf16)s[i];
}

// ---------------------------------------------------------------------------
// C(MxN,f32) = A(MxK,bf16) * B(NxK,bf16)^T   (both operands K-contiguous)
// Block tile 128x64, 8 waves, each wave: 16 rows x 4 WMMA n-subtiles.
// Double-buffered LDS, filled with async global->LDS B128 copies.
// ---------------------------------------------------------------------------
#define BM  128
#define BN  64
#define BKD 64

__global__ __launch_bounds__(256)
void gemm_bf16_nt(const __bf16* __restrict__ A, const __bf16* __restrict__ Bm,
                  float* __restrict__ C, int M, int N, int K) {
  __shared__ __bf16 sA[2][BM][BKD];   // 32 KB
  __shared__ __bf16 sB[2][BN][BKD];   // 16 KB

  const int tid  = threadIdx.x;
  const int wave = tid >> 5;
  const int lane = tid & 31;
  const int tileM = blockIdx.y * BM;
  const int tileN = blockIdx.x * BN;
  const int n16  = lane & 15;
  const int half = lane >> 4;
  const int kb8  = half * 8;
  const int kb16 = half * 16;
  const int mo   = half * 8;

  // async fill of one K-slab (128x64 A, 64x64 B): 16B per async op per lane
  auto issue_stage = [&](int buf, int k0) {
    const int ra = tid >> 1, ca = (tid & 1) * 32;          // A: 32 elems/thread
    const __bf16* ga = A + (size_t)(tileM + ra) * K + k0 + ca;
    #pragma unroll
    for (int j = 0; j < 4; ++j)
      async_copy_b128((unsigned)(size_t)&sA[buf][ra][ca + j * 8], ga + j * 8);
    const int rb = tid >> 2, cb = (tid & 3) * 16;          // B: 16 elems/thread
    const __bf16* gb = Bm + (size_t)(tileN + rb) * K + k0 + cb;
    #pragma unroll
    for (int j = 0; j < 2; ++j)
      async_copy_b128((unsigned)(size_t)&sB[buf][rb][cb + j * 8], gb + j * 8);
  };

  v8f acc[4] = {};

  issue_stage(0, 0);
  int stage = 0;
  for (int k0 = 0; k0 < K; k0 += BKD, stage ^= 1) {
    wait_async0();        // own wave's async fill of `stage` complete
    __syncthreads();      // -> every wave's fill complete
    if (k0 + BKD < K) issue_stage(stage ^ 1, k0 + BKD);  // prefetch next slab

    #pragma unroll
    for (int kk = 0; kk < BKD; kk += 32) {
      // A fragment (16x32): lanes 0-15 hold K 0..15, 16-31 K 16..31
      v16bf af;
      {
        const __bf16* p = &sA[stage][wave * 16 + n16][kk];
        #pragma unroll
        for (int j = 0; j < 8; ++j) {
          af[j]     = p[kb16 + j];
          af[8 + j] = p[kb16 + 8 + j];
        }
      }
      #pragma unroll
      for (int nt = 0; nt < 4; ++nt) {
        // B fragment (32x16): lane = column N, lanes 0-15 K=0..15, 16-31 K=16..31
        v16bf bfr;
        const __bf16* p = &sB[stage][nt * 16 + n16][kk + kb16];
        #pragma unroll
        for (int j = 0; j < 16; ++j) bfr[j] = p[j];
        acc[nt] = __builtin_amdgcn_wmma_f32_16x16x32_bf16(
            false, af, false, bfr, (short)0, acc[nt], false, false);
      }
    }
    __syncthreads();      // all waves done reading `stage` before it refills
  }

  // C layout: VGPR v -> row = v + 8*half, col = lane%16
  #pragma unroll
  for (int nt = 0; nt < 4; ++nt)
    #pragma unroll
    for (int v = 0; v < 8; ++v)
      C[(size_t)(tileM + wave * 16 + mo + v) * N + tileN + nt * 16 + n16] = acc[nt][v];
}

// ---------------------------------------------------------------------------
// Fused RMS-norm + RoPE (+ optional per-head gain), one wave per 128-elem row.
// Row layout: src[((b*S+s)*HROW + h) * 128 + d], same for dst (bf16).
// ---------------------------------------------------------------------------
__global__ __launch_bounds__(256)
void qk_norm_rope(const float* __restrict__ src, __bf16* __restrict__ dst,
                  const float* __restrict__ gain, int HROW, int S) {
  const int wave = threadIdx.x >> 5, lane = threadIdx.x & 31;
  const long ridx = (long)blockIdx.x * 8 + wave;
  const int h = (int)(ridx % HROW);
  const int s = (int)((ridx / HROW) % S);
  const float* row = src + ridx * 128;

  float a0 = row[lane * 2],      a1 = row[lane * 2 + 1];
  float b0 = row[64 + lane * 2], b1 = row[64 + lane * 2 + 1];
  float ss = a0 * a0 + a1 * a1 + b0 * b0 + b1 * b1;
  #pragma unroll
  for (int off = 16; off > 0; off >>= 1) ss += __shfl_xor(ss, off, 32);
  const float rstd = rsqrtf(ss * (1.0f / 128.0f) + 1.1920928955078125e-07f);
  a0 *= rstd; a1 *= rstd; b0 *= rstd; b1 *= rstd;

  const float g = gain ? gain[h] : 1.0f;
  const float i0 = (float)(lane * 2), i1 = (float)(lane * 2 + 1);
  const float inv0 = __powf(10000.0f, -i0 * (1.0f / 64.0f));
  const float inv1 = __powf(10000.0f, -i1 * (1.0f / 64.0f));
  const float p = (float)s;
  float c0, s0, c1, s1;
  __sincosf(p * inv0, &s0, &c0);
  __sincosf(p * inv1, &s1, &c1);

  __bf16* drow = dst + ridx * 128;
  drow[lane * 2]          = (__bf16)((a0 * c0 + b0 * s0) * g);
  drow[lane * 2 + 1]      = (__bf16)((a1 * c1 + b1 * s1) * g);
  drow[64 + lane * 2]     = (__bf16)((-a0 * s0 + b0 * c0) * g);
  drow[64 + lane * 2 + 1] = (__bf16)((-a1 * s1 + b1 * c1) * g);
}

// ---------------------------------------------------------------------------
// Flash attention, causal, GQA. Block = (b, h, 128 q-rows), 8 waves.
// qb: (B,S,H*hd) bf16; kb/vb: (B,S,KV*hd) bf16; of: (B,S,H*hd) f32
// K/V tiles streamed into LDS with async global->LDS copies.
// ---------------------------------------------------------------------------
__global__ __launch_bounds__(256)
void attn_flash(const __bf16* __restrict__ qb, const __bf16* __restrict__ kbuf,
                const __bf16* __restrict__ vbuf, float* __restrict__ of) {
  __shared__ __bf16 sK[64][128];      // 16 KB
  __shared__ __bf16 sV[64][128];      // 16 KB
  __shared__ __bf16 sP[8][16][64];    // 16 KB

  const int tid  = threadIdx.x;
  const int wave = tid >> 5;
  const int lane = tid & 31;
  const int qt  = blockIdx.x & 15;          // S/128 = 16 q tiles
  const int h   = (blockIdx.x >> 4) & 15;   // HEADS
  const int b   = blockIdx.x >> 8;
  const int kvh = h >> 2;                   // HEADS/KVH = 4
  const int qr0 = qt * 128 + wave * 16;
  const int n16  = lane & 15;
  const int half = lane >> 4;
  const int mo   = half * 8;
  const int kb8  = half * 8;
  const int kb16 = half * 16;

  // preload q fragments (K = head_dim = 128 -> 4 chunks of 32)
  v16bf qf[4];
  {
    const __bf16* qrow = qb + (size_t)(b * SEQ + qr0 + n16) * DIM + h * HD;
    #pragma unroll
    for (int ks = 0; ks < 4; ++ks)
      #pragma unroll
      for (int j = 0; j < 8; ++j) {
        qf[ks][j]     = qrow[ks * 32 + kb16 + j];
        qf[ks][8 + j] = qrow[ks * 32 + kb16 + 8 + j];
      }
  }

  float rmax[8], rsum[8];
  v8f oacc[8] = {};
  #pragma unroll
  for (int v = 0; v < 8; ++v) { rmax[v] = -1e30f; rsum[v] = 0.0f; }

  const float rscale = 0.08838834764831845f;  // 1/sqrt(128)
  const int nkt = (qt + 1) * 2;               // causal: kv tiles of 64

  for (int kt = 0; kt < nkt; ++kt) {
    { // async K/V tile fill, 64x128 each, 16B per op per lane
      int r = tid >> 2, c = (tid & 3) * 32;
      const __bf16* ksrc = kbuf + (size_t)(b * SEQ + kt * 64 + r) * KVD + kvh * HD + c;
      const __bf16* vsrc = vbuf + (size_t)(b * SEQ + kt * 64 + r) * KVD + kvh * HD + c;
      #pragma unroll
      for (int j = 0; j < 4; ++j) {
        async_copy_b128((unsigned)(size_t)&sK[r][c + j * 8], ksrc + j * 8);
        async_copy_b128((unsigned)(size_t)&sV[r][c + j * 8], vsrc + j * 8);
      }
    }
    wait_async0();
    __syncthreads();

    // S = q @ k^T : 16x64 scores per wave
    v8f sacc[4] = {};
    #pragma unroll
    for (int nt = 0; nt < 4; ++nt)
      #pragma unroll
      for (int ks = 0; ks < 4; ++ks) {
        v16bf bfr;  // B col n = kv row (nt*16+n16), contiguous head-dim
        const __bf16* p = &sK[nt * 16 + n16][ks * 32 + kb16];
        #pragma unroll
        for (int j = 0; j < 16; ++j) bfr[j] = p[j];
        sacc[nt] = __builtin_amdgcn_wmma_f32_16x16x32_bf16(
            false, qf[ks], false, bfr, (short)0, sacc[nt], false, false);
      }

    // scale + causal mask (row = qr0+mo+v, col = kt*64+nt*16+n16)
    #pragma unroll
    for (int nt = 0; nt < 4; ++nt)
      #pragma unroll
      for (int v = 0; v < 8; ++v) {
        float sc = sacc[nt][v] * rscale;
        sacc[nt][v] = ((kt * 64 + nt * 16 + n16) > (qr0 + mo + v)) ? -1e30f : sc;
      }

    // online softmax per row (xor-shuffles <=8 stay inside 16-lane halves)
    #pragma unroll
    for (int v = 0; v < 8; ++v) {
      float m = sacc[0][v];
      #pragma unroll
      for (int nt = 1; nt < 4; ++nt) m = fmaxf(m, sacc[nt][v]);
      #pragma unroll
      for (int off = 8; off > 0; off >>= 1) m = fmaxf(m, __shfl_xor(m, off, 32));
      const float nm = fmaxf(rmax[v], m);
      const float corr = __expf(rmax[v] - nm);
      rmax[v] = nm;
      float ls = 0.0f;
      #pragma unroll
      for (int nt = 0; nt < 4; ++nt) {
        float pp = __expf(sacc[nt][v] - nm);
        sacc[nt][v] = pp;
        ls += pp;
      }
      #pragma unroll
      for (int off = 8; off > 0; off >>= 1) ls += __shfl_xor(ls, off, 32);
      rsum[v] = rsum[v] * corr + ls;
      #pragma unroll
      for (int ns = 0; ns < 8; ++ns) oacc[ns][v] *= corr;
    }

    // P (bf16) to LDS so it can be re-read in A-fragment layout
    #pragma unroll
    for (int nt = 0; nt < 4; ++nt)
      #pragma unroll
      for (int v = 0; v < 8; ++v)
        sP[wave][mo + v][nt * 16 + n16] = (__bf16)sacc[nt][v];
    __syncthreads();

    // O += P @ V  (K = 64 kv positions -> 2 chunks of 32; N = 128 head dims)
    #pragma unroll
    for (int ks2 = 0; ks2 < 2; ++ks2) {
      v16bf pf;
      const __bf16* prow = &sP[wave][n16][ks2 * 32];
      #pragma unroll
      for (int j = 0; j < 8; ++j) {
        pf[j]     = prow[kb16 + j];
        pf[8 + j] = prow[kb16 + 8 + j];
      }
      #pragma unroll
      for (int ns = 0; ns < 8; ++ns) {
        v16bf vfr;  // B col n = head dim (ns*16+n16), rows = kv positions
        #pragma unroll
        for (int j = 0; j < 16; ++j)
          vfr[j] = sV[ks2 * 32 + kb16 + j][ns * 16 + n16];
        oacc[ns] = __builtin_amdgcn_wmma_f32_16x16x32_bf16(
            false, pf, false, vfr, (short)0, oacc[ns], false, false);
      }
    }
    __syncthreads();
  }

  // normalize + store f32
  #pragma unroll
  for (int v = 0; v < 8; ++v) {
    const float inv = 1.0f / rsum[v];
    float* orow = of + (size_t)(b * SEQ + qr0 + mo + v) * DIM + h * HD;
    #pragma unroll
    for (int ns = 0; ns < 8; ++ns)
      orow[ns * 16 + n16] = oacc[ns][v] * inv;
  }
}

// ---------------------------------------------------------------------------
// Host orchestration
// ---------------------------------------------------------------------------
extern "C" void kernel_launch(void* const* d_in, const int* in_sizes, int n_in,
                              void* d_out, int out_size, void* d_ws, size_t ws_size,
                              hipStream_t stream) {
  (void)in_sizes; (void)n_in; (void)out_size; (void)ws_size;
  const float* x      = (const float*)d_in[0];
  const float* Wq     = (const float*)d_in[1];
  const float* Wk     = (const float*)d_in[2];
  const float* Wv     = (const float*)d_in[3];
  const float* Wproj  = (const float*)d_in[4];
  const float* q_gain = (const float*)d_in[5];
  float* out = (float*)d_out;

  char* w = (char*)d_ws;
  size_t off = 0;
  auto alloc = [&](size_t bytes) {
    size_t o = off; off += (bytes + 255) & ~(size_t)255; return o;
  };
  __bf16* xb  = (__bf16*)(w + alloc((size_t)MROWS * DIM * 2));
  __bf16* Wqb = (__bf16*)(w + alloc((size_t)DIM * DIM * 2));
  __bf16* Wkb = (__bf16*)(w + alloc((size_t)KVD * DIM * 2));
  __bf16* Wvb = (__bf16*)(w + alloc((size_t)KVD * DIM * 2));
  __bf16* Wpb = (__bf16*)(w + alloc((size_t)DIM * DIM * 2));
  float*  Qf  = (float*) (w + alloc((size_t)MROWS * DIM * 4));
  float*  Kf  = (float*) (w + alloc((size_t)MROWS * KVD * 4));
  float*  Vf  = (float*) (w + alloc((size_t)MROWS * KVD * 4));
  __bf16* qbn = (__bf16*)(w + alloc((size_t)MROWS * DIM * 2));
  __bf16* kbn = (__bf16*)(w + alloc((size_t)MROWS * KVD * 2));
  __bf16* vbn = (__bf16*)(w + alloc((size_t)MROWS * KVD * 2));
  float*  of  = Qf;   // Qf consumed by qk_norm_rope before attention writes
  __bf16* yb  = xb;   // xb consumed by QKV GEMMs before y cast

  // 1) casts to bf16
  cast_f32_bf16<<<4096, 256, 0, stream>>>(x,     xb,  (long)MROWS * DIM);
  cast_f32_bf16<<<4096, 256, 0, stream>>>(Wq,    Wqb, (long)DIM * DIM);
  cast_f32_bf16<<<2048, 256, 0, stream>>>(Wk,    Wkb, (long)KVD * DIM);
  cast_f32_bf16<<<2048, 256, 0, stream>>>(Wv,    Wvb, (long)KVD * DIM);
  cast_f32_bf16<<<4096, 256, 0, stream>>>(Wproj, Wpb, (long)DIM * DIM);

  // 2) QKV projections (C = A * B^T)
  gemm_bf16_nt<<<dim3(DIM / BN, MROWS / BM), 256, 0, stream>>>(xb, Wqb, Qf, MROWS, DIM, DIM);
  gemm_bf16_nt<<<dim3(KVD / BN, MROWS / BM), 256, 0, stream>>>(xb, Wkb, Kf, MROWS, KVD, DIM);
  gemm_bf16_nt<<<dim3(KVD / BN, MROWS / BM), 256, 0, stream>>>(xb, Wvb, Vf, MROWS, KVD, DIM);

  // 3) fused RMS-norm + RoPE (+gain for q), and v cast
  qk_norm_rope<<<(MROWS * HEADS) / 8, 256, 0, stream>>>(Qf, qbn, q_gain, HEADS, SEQ);
  qk_norm_rope<<<(MROWS * KVH) / 8,   256, 0, stream>>>(Kf, kbn, nullptr, KVH, SEQ);
  cast_f32_bf16<<<2048, 256, 0, stream>>>(Vf, vbn, (long)MROWS * KVD);

  // 4) causal flash attention
  attn_flash<<<BATCH * HEADS * (SEQ / 128), 256, 0, stream>>>(qbn, kbn, vbn, of);

  // 5) output projection
  cast_f32_bf16<<<4096, 256, 0, stream>>>(of, yb, (long)MROWS * DIM);
  gemm_bf16_nt<<<dim3(DIM / BN, MROWS / BM), 256, 0, stream>>>(yb, Wpb, out, MROWS, DIM, DIM);
}
